// MultiQueryAttention_53377853555360
// MI455X (gfx1250) — compile-verified, matrix operand order checked by reading, not run on hardware
//
#include <hip/hip_runtime.h>

typedef __attribute__((ext_vector_type(16))) _Float16     v16h;
typedef __attribute__((ext_vector_type(8)))  float        v8f;
typedef __attribute__((ext_vector_type(4)))  unsigned int u32x4;
typedef __attribute__((ext_vector_type(8)))  unsigned int u32x8;

#define B_  2
#define S_  2048
#define D_  1024
#define H_  16
#define DH_ 64

#define USE_TDM 1

__device__ __forceinline__ _Float16 cvt_h(float x)    { return (_Float16)x; }
__device__ __forceinline__ _Float16 cvt_h(_Float16 x) { return x; }

// ---------------------------------------------------------------------------
// DPP16 butterfly reduction across each 16-lane half (no LDS, no waits).
//   0xB1  = quad_perm [1,0,3,2]  -> lane ^ 1
//   0x4E  = quad_perm [2,3,0,1]  -> lane ^ 2
//   0x141 = row_half_mirror      -> lane ^ 7 within 8  (valid once quads merged)
//   0x140 = row_mirror           -> lane ^ 15 within 16-lane row
// ---------------------------------------------------------------------------
template <int CTRL>
__device__ __forceinline__ float dpp_bfly(float x) {
  return __int_as_float(
      __builtin_amdgcn_update_dpp(0, __float_as_int(x), CTRL, 0xF, 0xF, true));
}
__device__ __forceinline__ float red16_max(float x) {
  x = fmaxf(x, dpp_bfly<0xB1>(x));
  x = fmaxf(x, dpp_bfly<0x4E>(x));
  x = fmaxf(x, dpp_bfly<0x141>(x));
  x = fmaxf(x, dpp_bfly<0x140>(x));
  return x;
}
__device__ __forceinline__ float red16_sum(float x) {
  x += dpp_bfly<0xB1>(x);
  x += dpp_bfly<0x4E>(x);
  x += dpp_bfly<0x141>(x);
  x += dpp_bfly<0x140>(x);
  return x;
}

// Load one 16x32 f16 WMMA A/B fragment from a row-major tile.
// Layout per CDNA5 ISA 7.12.2: lanes 0-15 = rows 0-15 holding K {0..7, 16..23},
// lanes 16-31 = rows 0-15 holding K {8..15, 24..31}.
__device__ __forceinline__ v16h load_frag16x32(const _Float16* base, int stride, int lane) {
  const int half = lane >> 4;
  const int row  = lane & 15;
  const _Float16* p = base + row * stride + half * 8;
  v16h f;
#pragma unroll
  for (int i = 0; i < 8; ++i) f[i] = p[i];
#pragma unroll
  for (int i = 0; i < 8; ++i) f[8 + i] = p[16 + i];
  return f;
}

// ---------------------------------------------------------------------------
// TDM: stage one 64x64 f16 tile (row stride 64 elems in memory) into LDS with
// hardware padding that yields an LDS row stride of 72 halves (144 B):
//   pad_interval = 4 -> pad after every 32 DWORDs (= one 128 B tile row)
//   pad_amount   = 3 -> insert 4 DWORDs (= 8 halves) of padding
// Descriptor per CDNA5 ISA ch.8 (D# group0 + group1), issued from SGPR tuples.
// ---------------------------------------------------------------------------
__device__ __forceinline__ void tdm_load_tile64x64(const _Float16* gsrc,
                                                   unsigned lds_byte_addr) {
  unsigned long long ga = (unsigned long long)(const void*)gsrc;
  u32x4 g0;
  g0[0] = 1u;                                             // count=1, user D#
  g0[1] = lds_byte_addr;                                  // lds_addr
  g0[2] = (unsigned)(ga & 0xFFFFFFFFu);                   // global_addr[31:0]
  g0[3] = (unsigned)((ga >> 32) & 0x01FFFFFFu)            // global_addr[56:32]
        | (2u << 30);                                     // type=2 ("image")
  u32x8 g1;
  g1[0] = (1u << 16)                                      // data_size = 2 B
        | (1u << 20)                                      // pad_enable
        | (4u << 22)                                      // pad_interval: 32 DW
        | (3u << 25);                                     // pad_amount: 4 DW
  g1[1] = (64u   << 16);                                  // tensor_dim0 = 64
  g1[2] = (2048u << 16);                                  // tensor_dim1 = 2048
  g1[3] = (64u   << 16);                                  // tile_dim0 = 64
  g1[4] = 64u;                                            // tile_dim1 = 64
  g1[5] = 64u;                                            // tensor_dim0_stride
  g1[6] = 0u;
  g1[7] = 0u;
  u32x4 gz = {0u, 0u, 0u, 0u};                            // groups 2/3 unused
  asm volatile("tensor_load_to_lds %0, %1, %2, %3"
               :: "s"(g0), "s"(g1), "s"(gz), "s"(gz) : "memory");
}

// ---------------------------------------------------------------------------
// Tiled GEMM + bias (+ output scale): C = (A @ W + bias) * out_scale.
// BM=128, BN=64, BK=32; 256 threads = 8 waves (4 along M x 2 along N),
// each wave computes a 32x32 patch as 2x2 v_wmma_f32_16x16x32_f16 tiles.
// ---------------------------------------------------------------------------
template <typename TIN, typename TOUT>
__global__ __launch_bounds__(256) void gemm_bias_f16wmma(
    const TIN* __restrict__ A, const float* __restrict__ W,
    const float* __restrict__ bias, TOUT* __restrict__ C,
    int M, int N, int K, float out_scale)
{
  constexpr int BM = 128, BN = 64, BK = 32, LDA = 40;
  __shared__ _Float16 As[BM * LDA];   // [m][k] f16
  __shared__ _Float16 Wt[BN * LDA];   // transposed: [n][k] f16

  const int tid  = threadIdx.x;
  const int lane = tid & 31;
  const int wave = tid >> 5;
  const int wm   = wave & 3;
  const int wn   = wave >> 2;
  const int m0   = blockIdx.x * BM;
  const int n0   = blockIdx.y * BN;

  v8f acc[2][2] = {};

  for (int k0 = 0; k0 < K; k0 += BK) {
    // Stage A tile (convert to f16), coalesced 32-wide rows.
#pragma unroll
    for (int i = tid; i < BM * BK; i += 256) {
      int r = i >> 5, c = i & 31;
      As[r * LDA + c] = cvt_h(A[(size_t)(m0 + r) * K + k0 + c]);
    }
    // Stage W tile transposed into [n][k].
#pragma unroll
    for (int i = tid; i < BK * BN; i += 256) {
      int k = i >> 6, n = i & 63;
      Wt[n * LDA + k] = cvt_h(W[(size_t)(k0 + k) * N + n0 + n]);
    }
    if (k0 + BK < K)  // hint next A tile toward L2 (global_prefetch_b8)
      __builtin_prefetch(&A[(size_t)(m0 + (tid >> 1)) * K + k0 + BK], 0, 0);
    __syncthreads();

    v16h a[2], b[2];
#pragma unroll
    for (int t = 0; t < 2; ++t)
      a[t] = load_frag16x32(&As[(wm * 32 + t * 16) * LDA], LDA, lane);
#pragma unroll
    for (int t = 0; t < 2; ++t)
      b[t] = load_frag16x32(&Wt[(wn * 32 + t * 16) * LDA], LDA, lane);

#pragma unroll
    for (int tm = 0; tm < 2; ++tm)
#pragma unroll
      for (int tn = 0; tn < 2; ++tn)
        acc[tm][tn] = __builtin_amdgcn_wmma_f32_16x16x32_f16(
            false, a[tm], false, b[tn], (short)0, acc[tm][tn], false, false);
    __syncthreads();
  }

  // Epilogue. C/D layout: lane = column (0-15 per half), VGPR r = row (+8 upper half).
  const int rbase = (lane >> 4) * 8;
  const int col_l = lane & 15;
#pragma unroll
  for (int tm = 0; tm < 2; ++tm)
#pragma unroll
    for (int tn = 0; tn < 2; ++tn) {
      int n = n0 + wn * 32 + tn * 16 + col_l;
      float bv = bias[n];
#pragma unroll
      for (int r = 0; r < 8; ++r) {
        int m = m0 + wm * 32 + tm * 16 + rbase + r;
        C[(size_t)m * N + n] = (TOUT)((acc[tm][tn][r] + bv) * out_scale);
      }
    }
}

// ---------------------------------------------------------------------------
// Flash MQA core. Grid: (S/64, B*H); 128 threads = 4 waves, 16 q-rows/wave.
// Q is pre-scaled by 1/sqrt(dh) in its projection. K tile staged by the TDM
// (hardware-padded to stride 72), V tile staged transposed via b128 loads.
// Softmax row reductions use DPP16 butterflies (co-execute with XDL WMMA).
// ---------------------------------------------------------------------------
__global__ __launch_bounds__(128) void mqa_flash_kernel(
    const _Float16* __restrict__ Q,   // [B,S,D]  (H*DH packed along D)
    const _Float16* __restrict__ Kb,  // [B,S,DH]
    const _Float16* __restrict__ Vb,  // [B,S,DH]
    _Float16* __restrict__ ctx)       // [B,S,D]
{
  constexpr int LDK = 72;
  __shared__ _Float16 Ks[64 * LDK];      // [key][dh]   (TDM destination)
  __shared__ _Float16 Vt[DH_ * LDK];     // [dh][key]
  __shared__ _Float16 Pb[4 * 16 * LDK];  // per-wave P tile, [q][key]

  const int tid  = threadIdx.x;
  const int lane = tid & 31;
  const int wave = tid >> 5;
  const int b    = blockIdx.y / H_;
  const int h    = blockIdx.y % H_;
  const int q0   = blockIdx.x * 64 + wave * 16;

  // Q fragments for this wave's 16 rows, dh split into two K=32 chunks.
  const _Float16* qg = Q + (size_t)(b * S_ + q0) * D_ + h * DH_;
  v16h aq[2];
#pragma unroll
  for (int kc = 0; kc < 2; ++kc)
    aq[kc] = load_frag16x32(qg + kc * 32, D_, lane);

  float m_run[8], l_run[8];
#pragma unroll
  for (int r = 0; r < 8; ++r) { m_run[r] = -3.0e38f; l_run[r] = 0.0f; }
  v8f o[4] = {};

  _Float16* pb = &Pb[wave * 16 * LDK];
  const unsigned ks_lds = (unsigned)(unsigned long long)(void*)&Ks[0];

  for (int s0 = 0; s0 < S_; s0 += 64) {
    const _Float16* kg = Kb + (size_t)(b * S_ + s0) * DH_;
    const _Float16* vg = Vb + (size_t)(b * S_ + s0) * DH_;

#if USE_TDM
    if (wave == 0)
      tdm_load_tile64x64(kg, ks_lds);            // K tile -> Ks (TENSORcnt)
#else
#pragma unroll
    for (int i = tid; i < 64 * 8; i += 128) {
      int kk = i >> 3, c8 = (i & 7) * 8;
      const uint4 pk = ((const uint4*)kg)[i];
      const _Float16* pf = (const _Float16*)&pk;
#pragma unroll
      for (int j = 0; j < 8; ++j) Ks[kk * LDK + c8 + j] = pf[j];
    }
#endif
    // V tile staged transposed: b128 global loads, scatter ds stores.
#pragma unroll
    for (int i = tid; i < 64 * 8; i += 128) {
      int kk = i >> 3, c8 = (i & 7) * 8;
      const uint4 pk = ((const uint4*)vg)[i];
      const _Float16* pf = (const _Float16*)&pk;
#pragma unroll
      for (int j = 0; j < 8; ++j) Vt[(c8 + j) * LDK + kk] = pf[j];
    }
#if USE_TDM
    if (wave == 0)
      __builtin_amdgcn_s_wait_tensorcnt(0);      // drain TDM before barrier
#endif
    __syncthreads();

    // Scores: 4 key-tiles of 16, chained over dh (2 x K=32). Scale pre-folded.
    v8f sc[4] = {};
#pragma unroll
    for (int t = 0; t < 4; ++t) {
#pragma unroll
      for (int kc = 0; kc < 2; ++kc) {
        v16h bk = load_frag16x32(&Ks[(t * 16) * LDK + kc * 32], LDK, lane);
        sc[t] = __builtin_amdgcn_wmma_f32_16x16x32_f16(
            false, aq[kc], false, bk, (short)0, sc[t], false, false);
      }
    }

    // Online softmax: row r lives in VGPR r across one 16-lane half.
    float mnew[8], alpha[8];
#pragma unroll
    for (int r = 0; r < 8; ++r) {
      float mt = fmaxf(fmaxf(sc[0][r], sc[1][r]), fmaxf(sc[2][r], sc[3][r]));
      mt = red16_max(mt);
      mnew[r]  = fmaxf(m_run[r], mt);
      alpha[r] = __expf(m_run[r] - mnew[r]);
      m_run[r] = mnew[r];
    }
#pragma unroll
    for (int t = 0; t < 4; ++t)
#pragma unroll
      for (int r = 0; r < 8; ++r)
        sc[t][r] = __expf(sc[t][r] - mnew[r]);
#pragma unroll
    for (int r = 0; r < 8; ++r) {
      float ls = red16_sum(sc[0][r] + sc[1][r] + sc[2][r] + sc[3][r]);
      l_run[r] = l_run[r] * alpha[r] + ls;
    }
#pragma unroll
    for (int t = 0; t < 4; ++t)
#pragma unroll
      for (int r = 0; r < 8; ++r)
        o[t][r] *= alpha[r];

    // Re-layout P (C-layout -> row-major f16) through this wave's LDS tile.
    {
      const int rb = (lane >> 4) * 8, cl = lane & 15;
#pragma unroll
      for (int t = 0; t < 4; ++t)
#pragma unroll
        for (int r = 0; r < 8; ++r)
          pb[(rb + r) * LDK + t * 16 + cl] = (_Float16)sc[t][r];
    }

    // O += P @ V : 4 dh-tiles, key dim 64 as two K=32 chunks.
#pragma unroll
    for (int kc = 0; kc < 2; ++kc) {
      v16h ap = load_frag16x32(pb + kc * 32, LDK, lane);
#pragma unroll
      for (int t = 0; t < 4; ++t) {
        v16h bv = load_frag16x32(&Vt[(t * 16) * LDK + kc * 32], LDK, lane);
        o[t] = __builtin_amdgcn_wmma_f32_16x16x32_f16(
            false, ap, false, bv, (short)0, o[t], false, false);
      }
    }
    __syncthreads();
  }

  // Normalize and store ctx (f16, [B,S,D]).
  const int rb = (lane >> 4) * 8, cl = lane & 15;
#pragma unroll
  for (int r = 0; r < 8; ++r) {
    float inv = 1.0f / l_run[r];
    int q = q0 + rb + r;
    _Float16* cg = ctx + (size_t)(b * S_ + q) * D_ + h * DH_;
#pragma unroll
    for (int t = 0; t < 4; ++t)
      cg[t * 16 + cl] = (_Float16)(o[t][r] * inv);
  }
}

// ---------------------------------------------------------------------------
extern "C" void kernel_launch(void* const* d_in, const int* in_sizes, int n_in,
                              void* d_out, int out_size, void* d_ws, size_t ws_size,
                              hipStream_t stream) {
  (void)in_sizes; (void)n_in; (void)out_size; (void)ws_size;
  const float* q   = (const float*)d_in[0];
  const float* k   = (const float*)d_in[1];
  const float* v   = (const float*)d_in[2];
  const float* w_q = (const float*)d_in[3];
  const float* b_q = (const float*)d_in[4];
  const float* w_k = (const float*)d_in[5];
  const float* b_k = (const float*)d_in[6];
  const float* w_v = (const float*)d_in[7];
  const float* b_v = (const float*)d_in[8];
  const float* w_o = (const float*)d_in[9];
  const float* b_o = (const float*)d_in[10];
  float* out = (float*)d_out;

  const int M = B_ * S_;  // 4096
  char* ws = (char*)d_ws;
  _Float16* Qb = (_Float16*)ws;  ws += (size_t)M * D_  * sizeof(_Float16);
  _Float16* Kf = (_Float16*)ws;  ws += (size_t)M * DH_ * sizeof(_Float16);
  _Float16* Vf = (_Float16*)ws;  ws += (size_t)M * DH_ * sizeof(_Float16);
  _Float16* Cx = (_Float16*)ws;

  const float qscale = 0.125f;  // 1/sqrt(DH), folded into Q projection

  dim3 blk(256);
  gemm_bias_f16wmma<float, _Float16>
      <<<dim3(M / 128, D_ / 64), blk, 0, stream>>>(q, w_q, b_q, Qb, M, D_, D_, qscale);
  gemm_bias_f16wmma<float, _Float16>
      <<<dim3(M / 128, 1), blk, 0, stream>>>(k, w_k, b_k, Kf, M, DH_, D_, 1.0f);
  gemm_bias_f16wmma<float, _Float16>
      <<<dim3(M / 128, 1), blk, 0, stream>>>(v, w_v, b_v, Vf, M, DH_, D_, 1.0f);

  mqa_flash_kernel<<<dim3(S_ / 64, B_ * H_), dim3(128), 0, stream>>>(Qb, Kf, Vf, Cx);

  gemm_bias_f16wmma<_Float16, float>
      <<<dim3(M / 128, D_ / 64), blk, 0, stream>>>(Cx, w_o, b_o, out, M, D_, D_, 1.0f);
}
